// Lowrank_Spattention_47021301956896
// MI455X (gfx1250) — matrix operand order repaired and checked by reading, not compiled, not run on hardware
//
#include <hip/hip_runtime.h>

// ---------------------------------------------------------------------------
// Low-rank pooled attention for MI455X (gfx1250), wave32 + WMMA bf16.
// B=8, N=8192, DIM=256, HEAD=4, RANK=64, HEAD_DIM=64.
// ---------------------------------------------------------------------------

typedef __attribute__((ext_vector_type(16))) __bf16 v16bf;
typedef __attribute__((ext_vector_type(8)))  __bf16 v8bf;
typedef __attribute__((ext_vector_type(8)))  float  v8f;

#define DEV __device__ __forceinline__

// Build a 16-element bf16 A/B fragment from two 16-byte contiguous LDS runs.
DEV v16bf frag16(const __bf16* p0, const __bf16* p1) {
  v8bf lo = *(const v8bf*)p0;
  v8bf hi = *(const v8bf*)p1;
  return __builtin_shufflevector(lo, hi, 0,1,2,3,4,5,6,7,8,9,10,11,12,13,14,15);
}

DEV v8f wmma_bf16(v16bf a, v16bf b, v8f c) {
  // D = A(16x32) * B(32x16) + C, f32 accumulate
  return __builtin_amdgcn_wmma_f32_16x16x32_bf16(false, a, false, b, (short)0, c,
                                                 false, false);
}

// ---------------------------------------------------------------------------
// Kernel 1: xv = x @ Wv + bv   (65536x256 @ 256x256), written to d_out.
// Weights staged transposed (bf16) in LDS, 128 output cols at a time.
// 256 threads = 8 waves: 4 M-waves x 2 N-waves, each wave 16 rows x 64 cols.
// ---------------------------------------------------------------------------
__global__ void lrsa_gemm_xv(const float* __restrict__ X, const float* __restrict__ W,
                             const float* __restrict__ bias, float* __restrict__ out) {
  extern __shared__ char smem[];
  __bf16* Wt = (__bf16*)smem;               // [128][264] bf16 = 67584 B (Wt[n][k])
  __bf16* As = (__bf16*)(smem + 67584);     // [64][264]  bf16 = 33792 B (As[m][k])
  const int tid = threadIdx.x;
  const int wave = tid >> 5, lane = tid & 31, hf = lane >> 4, m15 = lane & 15;
  const int wm = wave & 3, wn = wave >> 2;

  for (int ch = 0; ch < 2; ++ch) {
    const int nh0 = ch * 128;
    __syncthreads();
    // stage transposed weight half as bf16
    for (int i = tid; i < 256 * 128; i += 256) {
      int k = i >> 7, nl = i & 127;
      Wt[nl * 264 + k] = (__bf16)W[k * 256 + nh0 + nl];
    }
    __syncthreads();
    for (int t = blockIdx.x; t < 1024; t += gridDim.x) {
      const int b = t >> 7, n0 = (t & 127) * 64;
      const size_t row0 = (size_t)b * 8192 + n0;
      __syncthreads();
      // stage 64x256 input tile as bf16
      const float4* X4 = (const float4*)(X + row0 * 256);
      for (int i = tid; i < 64 * 64; i += 256) {
        int r = i >> 6, c4 = i & 63;
        float4 v = X4[r * 64 + c4];
        __bf16* d = &As[r * 264 + c4 * 4];
        d[0] = (__bf16)v.x; d[1] = (__bf16)v.y; d[2] = (__bf16)v.z; d[3] = (__bf16)v.w;
      }
      __syncthreads();
      v8f acc[4] = {};
      for (int k0 = 0; k0 < 256; k0 += 32) {
        const __bf16* ap = &As[(wm * 16 + m15) * 264 + k0 + hf * 16];
        v16bf a = frag16(ap, ap + 8);
#pragma unroll
        for (int ct = 0; ct < 4; ++ct) {
          const __bf16* bp = &Wt[(wn * 64 + ct * 16 + m15) * 264 + k0 + hf * 16];
          acc[ct] = wmma_bf16(a, frag16(bp, bp + 8), acc[ct]);
        }
      }
#pragma unroll
      for (int ct = 0; ct < 4; ++ct) {
        int c = nh0 + wn * 64 + ct * 16 + m15;
        float bc = bias[c];
#pragma unroll
        for (int v = 0; v < 8; ++v) {
          int r = wm * 16 + v + 8 * hf;
          out[(row0 + r) * 256 + c] = acc[ct][v] + bc;
        }
      }
    }
  }
}

// ---------------------------------------------------------------------------
// Kernel 2: q = z @ Wq + bq, then s[b,n,h,r] = (q_h . K_h^T) / 8  -> ws.s
// Two chained WMMA stages per tile; K latents staged [h][r][d] bf16 in LDS.
// ---------------------------------------------------------------------------
__global__ void lrsa_gemm_q_attn(const float* __restrict__ Z, const float* __restrict__ Wq,
                                 const float* __restrict__ bq, const float* __restrict__ Kl,
                                 float* __restrict__ s_out) {
  extern __shared__ char smem[];
  __bf16* Wt = (__bf16*)smem;                         // [128][264] = 67584 B
  __bf16* As = (__bf16*)(smem + 67584);               // [64][264]  = 33792 B
  __bf16* Qs = (__bf16*)(smem + 67584 + 33792);       // [64][136]  = 17408 B
  __bf16* Kd = (__bf16*)(smem + 67584 + 33792 + 17408); // [2][64][72] = 18432 B
  const int tid = threadIdx.x;
  const int wave = tid >> 5, lane = tid & 31, hf = lane >> 4, m15 = lane & 15;
  const int wm = wave & 3, wn = wave >> 2;

  for (int ch = 0; ch < 2; ++ch) {
    const int nh0 = ch * 128;
    __syncthreads();
    for (int i = tid; i < 256 * 128; i += 256) {
      int k = i >> 7, nl = i & 127;
      Wt[nl * 264 + k] = (__bf16)Wq[k * 256 + nh0 + nl];
    }
    // K[r][h][d] -> Kd[hl][r][d] for heads {2ch, 2ch+1}
    for (int i = tid; i < 2 * 64 * 64; i += 256) {
      int hl = i >> 12, rr = (i >> 6) & 63, d = i & 63;
      Kd[(hl * 64 + rr) * 72 + d] = (__bf16)Kl[(rr * 4 + (ch * 2 + hl)) * 64 + d];
    }
    __syncthreads();
    for (int t = blockIdx.x; t < 1024; t += gridDim.x) {
      const int b = t >> 7, n0 = (t & 127) * 64;
      const size_t row0 = (size_t)b * 8192 + n0;
      __syncthreads();
      const float4* Z4 = (const float4*)(Z + row0 * 256);
      for (int i = tid; i < 64 * 64; i += 256) {
        int r = i >> 6, c4 = i & 63;
        float4 v = Z4[r * 64 + c4];
        __bf16* d = &As[r * 264 + c4 * 4];
        d[0] = (__bf16)v.x; d[1] = (__bf16)v.y; d[2] = (__bf16)v.z; d[3] = (__bf16)v.w;
      }
      __syncthreads();
      // ---- GEMM1: q half (64 x 128) ----
      v8f acc[4] = {};
      for (int k0 = 0; k0 < 256; k0 += 32) {
        const __bf16* ap = &As[(wm * 16 + m15) * 264 + k0 + hf * 16];
        v16bf a = frag16(ap, ap + 8);
#pragma unroll
        for (int ct = 0; ct < 4; ++ct) {
          const __bf16* bp = &Wt[(wn * 64 + ct * 16 + m15) * 264 + k0 + hf * 16];
          acc[ct] = wmma_bf16(a, frag16(bp, bp + 8), acc[ct]);
        }
      }
      // q + bias -> LDS (bf16)
#pragma unroll
      for (int ct = 0; ct < 4; ++ct) {
        int cl = wn * 64 + ct * 16 + m15;
        float bc = bq[nh0 + cl];
#pragma unroll
        for (int v = 0; v < 8; ++v) {
          int r = wm * 16 + v + 8 * hf;
          Qs[r * 136 + cl] = (__bf16)(acc[ct][v] + bc);
        }
      }
      __syncthreads();
      // ---- GEMM2: logits, wave wn handles head_local = wn (64 r-cols) ----
      v8f sa[4] = {};
      for (int k0 = 0; k0 < 64; k0 += 32) {
        const __bf16* ap = &Qs[(wm * 16 + m15) * 136 + wn * 64 + k0 + hf * 16];
        v16bf a = frag16(ap, ap + 8);
#pragma unroll
        for (int rt = 0; rt < 4; ++rt) {
          const __bf16* bp = &Kd[(wn * 64 + rt * 16 + m15) * 72 + k0 + hf * 16];
          sa[rt] = wmma_bf16(a, frag16(bp, bp + 8), sa[rt]);
        }
      }
      const int h = ch * 2 + wn;
#pragma unroll
      for (int rt = 0; rt < 4; ++rt)
#pragma unroll
        for (int v = 0; v < 8; ++v) {
          int r = wm * 16 + v + 8 * hf;
          int rr = rt * 16 + m15;
          s_out[((row0 + r) * 4 + h) * 64 + rr] = sa[rt][v] * 0.125f;
        }
      __syncthreads();
    }
  }
}

// ---------------------------------------------------------------------------
// Kernel 3: column softmax stats over n: cmax/csum per (b,h,r).
// 32 WGs x 1024 threads; thread = (seg 0..15, r 0..63), two passes.
// ---------------------------------------------------------------------------
__global__ void lrsa_colstats(const float* __restrict__ s, float* __restrict__ cmax,
                              float* __restrict__ csum) {
  __shared__ float red[1024];
  const int bh = blockIdx.x;               // b*4 + h
  const int b = bh >> 2, h = bh & 3;
  const int tid = threadIdx.x;
  const int r = tid & 63, seg = tid >> 6;  // 16 segments of 512 rows
  const float* col = s + (size_t)b * 2097152 + h * 64 + r;
  float m = -3.4e38f;
  for (int n = seg * 512; n < (seg + 1) * 512; ++n)
    m = fmaxf(m, col[(size_t)n * 256]);
  red[tid] = m; __syncthreads();
  if (tid < 64) {
    float mm = red[tid];
#pragma unroll
    for (int j = 1; j < 16; ++j) mm = fmaxf(mm, red[tid + 64 * j]);
    red[tid] = mm;
    cmax[bh * 64 + tid] = mm;
  }
  __syncthreads();
  const float mm = red[r];
  __syncthreads();
  float sum = 0.f;
  for (int n = seg * 512; n < (seg + 1) * 512; ++n)
    sum += __expf(col[(size_t)n * 256] - mm);
  red[tid] = sum; __syncthreads();
  if (tid < 64) {
    float ss = red[tid];
#pragma unroll
    for (int j = 1; j < 16; ++j) ss += red[tid + 64 * j];
    csum[bh * 64 + tid] = ss;
  }
}

// ---------------------------------------------------------------------------
// Kernel 4: pooled partials. WG = (b,h,split): softmax over r on the fly,
// transposed-stage a1^T and xv^T bf16 in LDS, WMMA over n-chunks of 256.
// Deterministic split-K: each split writes its own 64x64 partial block.
// ---------------------------------------------------------------------------
__global__ void lrsa_pooled(const float* __restrict__ s, const float* __restrict__ xv,
                            float* __restrict__ ppart) {
  extern __shared__ char smem[];
  __bf16* a1T = (__bf16*)smem;             // [64][264] (a1T[r][n_local])
  __bf16* xvT = (__bf16*)(smem + 33792);   // [64][264] (xvT[d][n_local])
  const int tid = threadIdx.x;
  const int wave = tid >> 5, lane = tid & 31, hf = lane >> 4, m15 = lane & 15;
  const int wm = wave & 3, wn = wave >> 2;
  const int gid = blockIdx.x;              // (b*4+h)*8 + split
  const int split = gid & 7, bh = gid >> 3, h = bh & 3, b = bh >> 2;

  v8f acc[2] = {};
  for (int chunk = 0; chunk < 4; ++chunk) {
    __syncthreads();
    const size_t n = (size_t)split * 1024 + chunk * 256 + tid;
    const size_t rowg = (size_t)b * 8192 + n;
    // softmax over r for this row, write a1^T bf16
    float e[64];
    const float4* s4 = (const float4*)(s + rowg * 256 + h * 64);
    float mx = -3.4e38f;
#pragma unroll
    for (int j = 0; j < 16; ++j) {
      float4 v = s4[j];
      e[4 * j] = v.x; e[4 * j + 1] = v.y; e[4 * j + 2] = v.z; e[4 * j + 3] = v.w;
      mx = fmaxf(fmaxf(fmaxf(mx, v.x), fmaxf(v.y, v.z)), v.w);
    }
    float sum = 0.f;
#pragma unroll
    for (int j = 0; j < 64; ++j) { e[j] = __expf(e[j] - mx); sum += e[j]; }
    const float inv = 1.f / sum;
#pragma unroll
    for (int j = 0; j < 64; ++j) a1T[j * 264 + tid] = (__bf16)(e[j] * inv);
    // xv row -> xv^T bf16
    const float4* x4 = (const float4*)(xv + rowg * 256 + h * 64);
#pragma unroll
    for (int j = 0; j < 16; ++j) {
      float4 v = x4[j];
      xvT[(4 * j + 0) * 264 + tid] = (__bf16)v.x;
      xvT[(4 * j + 1) * 264 + tid] = (__bf16)v.y;
      xvT[(4 * j + 2) * 264 + tid] = (__bf16)v.z;
      xvT[(4 * j + 3) * 264 + tid] = (__bf16)v.w;
    }
    __syncthreads();
    // partial (64r x 64d) += a1^T(64 x 256) @ xv(256 x 64)
    for (int k0 = 0; k0 < 256; k0 += 32) {
      const __bf16* ap = &a1T[(wm * 16 + m15) * 264 + k0 + hf * 16];
      v16bf a = frag16(ap, ap + 8);
#pragma unroll
      for (int ct = 0; ct < 2; ++ct) {
        const __bf16* bp = &xvT[(wn * 32 + ct * 16 + m15) * 264 + k0 + hf * 16];
        acc[ct] = wmma_bf16(a, frag16(bp, bp + 8), acc[ct]);
      }
    }
  }
  float* pp = ppart + (size_t)split * 131072;
#pragma unroll
  for (int ct = 0; ct < 2; ++ct)
#pragma unroll
    for (int v = 0; v < 8; ++v) {
      int r = wm * 16 + v + 8 * hf;
      int d = wn * 32 + ct * 16 + m15;
      pp[(((size_t)b * 64 + r) * 4 + h) * 64 + d] = acc[ct][v];
    }
}

__global__ void lrsa_reduce_pooled(const float* __restrict__ pp, float* __restrict__ pooled) {
  int i = blockIdx.x * 256 + threadIdx.x;   // 131072 elements
  float acc = 0.f;
#pragma unroll
  for (int sp = 0; sp < 8; ++sp) acc += pp[(size_t)sp * 131072 + i];
  pooled[i] = acc;
}

// ---------------------------------------------------------------------------
// Kernel 5: v = softmax_n(s) @ pooled, out = sigm(alpha)*xv + sigm(beta)*v.
// xv lives in d_out already -> read/modify/write.
// ---------------------------------------------------------------------------
__global__ void lrsa_final(const float* __restrict__ s, const float* __restrict__ pooled,
                           const float* __restrict__ cmax, const float* __restrict__ csum,
                           const float* __restrict__ alpha, const float* __restrict__ beta,
                           float* __restrict__ out) {
  extern __shared__ char smem[];
  __bf16* a2s = (__bf16*)smem;              // [64][264]   = 33792 B
  __bf16* pT  = (__bf16*)(smem + 33792);    // [4][64][72] = 36864 B (pT[h][d][r])
  const int tid = threadIdx.x;
  const int wave = tid >> 5, lane = tid & 31, hf = lane >> 4, m15 = lane & 15;
  const int wm = wave & 3, wn = wave >> 2;
  const int t = blockIdx.x;                 // 1024 tiles
  const int b = t >> 7, n0 = (t & 127) * 64;
  const size_t row0 = (size_t)b * 8192 + n0;

  // stage pooled^T bf16
  for (int i = tid; i < 16384; i += 256) {
    int h = i >> 12, r = (i >> 6) & 63, d = i & 63;
    pT[(h * 64 + d) * 72 + r] = (__bf16)pooled[(((size_t)b * 64 + r) * 4 + h) * 64 + d];
  }
  // stage a2 tile: thread owns column c = tid (h*64+r), rows 0..63
  const float cm = cmax[b * 256 + tid];
  const float ci = 1.f / csum[b * 256 + tid];
  for (int nl = 0; nl < 64; ++nl) {
    float sv = s[(row0 + nl) * 256 + tid];
    a2s[nl * 264 + tid] = (__bf16)(__expf(sv - cm) * ci);
  }
  __syncthreads();

  v8f vacc[8] = {};
#pragma unroll
  for (int ct = 0; ct < 8; ++ct) {
    const int c0 = wn * 128 + ct * 16;
    const int h = c0 >> 6, d0 = c0 & 63;
    for (int k0 = 0; k0 < 64; k0 += 32) {
      const __bf16* ap = &a2s[(wm * 16 + m15) * 264 + h * 64 + k0 + hf * 16];
      v16bf a = frag16(ap, ap + 8);
      const __bf16* bp = &pT[(h * 64 + d0 + m15) * 72 + k0 + hf * 16];
      vacc[ct] = wmma_bf16(a, frag16(bp, bp + 8), vacc[ct]);
    }
  }
#pragma unroll
  for (int ct = 0; ct < 8; ++ct) {
    const int c0 = wn * 128 + ct * 16;
    const int h = c0 >> 6;
    const int c = c0 + m15;
    const float ga = 1.f / (1.f + __expf(-alpha[h]));
    const float gb = 1.f / (1.f + __expf(-beta[h]));
#pragma unroll
    for (int v = 0; v < 8; ++v) {
      int r = wm * 16 + v + 8 * hf;
      size_t idx = (row0 + r) * 256 + c;
      out[idx] = ga * out[idx] + gb * vacc[ct][v];
    }
  }
}

// ---------------------------------------------------------------------------
extern "C" void kernel_launch(void* const* d_in, const int* in_sizes, int n_in,
                              void* d_out, int out_size, void* d_ws, size_t ws_size,
                              hipStream_t stream) {
  const float* x     = (const float*)d_in[0];
  const float* z     = (const float*)d_in[1];
  const float* Wq    = (const float*)d_in[2];
  const float* bq    = (const float*)d_in[3];
  const float* Kl    = (const float*)d_in[4];
  const float* Wv    = (const float*)d_in[5];
  const float* bv    = (const float*)d_in[6];
  const float* alpha = (const float*)d_in[7];
  const float* beta  = (const float*)d_in[8];
  float* out = (float*)d_out;

  float* ws     = (float*)d_ws;
  float* s      = ws;                       // 16,777,216 f32 (64 MB)
  float* cmaxp  = ws + 16777216;            // 2048
  float* csump  = cmaxp + 2048;             // 2048
  float* ppart  = csump + 2048;             // 8 * 131072
  float* pooled = ppart + 8 * 131072;       // 131072   (total ~71.8 MB)

  lrsa_gemm_xv<<<256, 256, 101376, stream>>>(x, Wv, bv, out);
  lrsa_gemm_q_attn<<<256, 256, 137216, stream>>>(z, Wq, bq, Kl, s);
  lrsa_colstats<<<32, 1024, 0, stream>>>(s, cmaxp, csump);
  lrsa_pooled<<<256, 256, 67584, stream>>>(s, out, ppart);
  lrsa_reduce_pooled<<<512, 256, 0, stream>>>(ppart, pooled);
  lrsa_final<<<1024, 256, 70656, stream>>>(s, pooled, cmaxp, csump, alpha, beta, out);
}
